// Sampler_65120294142321
// MI455X (gfx1250) — compile-verified, matrix operand order checked by reading, not run on hardware
//
#include <hip/hip_runtime.h>
#include <hip/hip_bf16.h>
#include <limits.h>

#define KTOP     64
#define NEG_INF  (-1.0e30f)
#define TPB1     256
#define CH       1024        /* floats per streamed chunk: 256 thr * 4 */
#define NBINS    4096
#define CAND_MAX 2048

typedef __attribute__((ext_vector_type(16))) _Float16 v16h;
typedef __attribute__((ext_vector_type(8)))  float    v8f;
typedef __attribute__((ext_vector_type(4)))  float    f4;
typedef __attribute__((ext_vector_type(4)))  int      v4i;

#if defined(__HIP_DEVICE_COMPILE__) && __has_builtin(__builtin_amdgcn_global_load_async_to_lds_b128)
#define USE_ASYNC 1
#else
#define USE_ASYNC 0
#endif

// Monotonic unsigned key: a > b  <=>  fkey(a) > fkey(b)
__device__ __forceinline__ unsigned fkey(float f) {
    unsigned u = __float_as_uint(f);
    return (u & 0x80000000u) ? ~u : (u | 0x80000000u);
}

#if USE_ASYNC
// Builtin signature (from hipcc diagnostic): arg0 = int4 in global (AS1),
// arg1 = LDS side, args 2/3 = immediate offset / cache-policy.
typedef __attribute__((address_space(1))) v4i* gv4i_p;
typedef __attribute__((address_space(3))) v4i* lv4i_p;

__device__ __forceinline__ void async_copy16(const float* g, float* l) {
    // 16 bytes/lane, one async instruction per wave -> ASYNCcnt++
    __builtin_amdgcn_global_load_async_to_lds_b128((gv4i_p)g, (lv4i_p)l, 0, 0);
}
__device__ __forceinline__ void async_wait1() {
#if __has_builtin(__builtin_amdgcn_s_wait_asynccnt)
    __builtin_amdgcn_s_wait_asynccnt(1);
#else
    asm volatile("s_wait_asynccnt 0x1" ::: "memory");
#endif
}
__device__ __forceinline__ void async_wait0() {
#if __has_builtin(__builtin_amdgcn_s_wait_asynccnt)
    __builtin_amdgcn_s_wait_asynccnt(0);
#else
    asm volatile("s_wait_asynccnt 0x0" ::: "memory");
#endif
}
#endif

// ---------------------------------------------------------------------------
// Kernel 1: per row -> logsumexp, top-64 values (scaled by 1/temp) + indices.
// Pass A: stream row (async double-buffer through LDS), online LSE + 12-bit
//         key histogram. Pass B: re-stream (L2-resident), collect candidates
//         above histogram threshold, then exact top-64 selection in LDS.
// ---------------------------------------------------------------------------
__global__ void __launch_bounds__(TPB1)
topk_lse_kernel(const float* __restrict__ logits,
                const float* __restrict__ temperature,
                float* __restrict__ ws_vals,
                int*   __restrict__ ws_idx,
                float* __restrict__ ws_logZ,
                int V)
{
    const int row = blockIdx.x;
    const int tid = threadIdx.x;
    const float* rowp = logits + (size_t)row * (size_t)V;

    __shared__ __align__(16) float tile[2][CH];
    __shared__ unsigned hist[NBINS];
    __shared__ float    cval[CAND_MAX];
    __shared__ int      cidx[CAND_MAX];
    __shared__ unsigned ccount;
    __shared__ unsigned s_b0;
    __shared__ float    red_v[TPB1];
    __shared__ float    red_s[TPB1];
    __shared__ int      red_i[TPB1];
    __shared__ int      red_p[TPB1];

    for (int b = tid; b < NBINS; b += TPB1) hist[b] = 0u;
    if (tid == 0) ccount = 0u;
    __syncthreads();

    const int nFull = V / CH;

    // ---------------- Pass A: online logsumexp + histogram ----------------
    float m = -3.4e38f, s = 0.0f;
#define PROC_A(x)                                              \
    do {                                                       \
        float _x = (x);                                        \
        if (_x <= m) { s += __expf(_x - m); }                  \
        else         { s = s * __expf(m - _x) + 1.0f; m = _x; }\
        atomicAdd(&hist[fkey(_x) >> 20], 1u);                  \
    } while (0)

#if USE_ASYNC
    if (nFull > 0) {
        async_copy16(rowp + tid * 4, &tile[0][tid * 4]);
        for (int it = 0; it < nFull; ++it) {
            if (it + 1 < nFull) {
                async_copy16(rowp + (size_t)(it + 1) * CH + tid * 4,
                             &tile[(it + 1) & 1][tid * 4]);
                async_wait1();
            } else {
                async_wait0();
            }
            __syncthreads();
            const float* tp = tile[it & 1] + tid * 4;
            #pragma unroll
            for (int k = 0; k < 4; ++k) PROC_A(tp[k]);
            __syncthreads();
        }
    }
#else
    for (int it = 0; it < nFull; ++it) {
        f4 v = *(const f4*)(rowp + (size_t)it * CH + tid * 4);
        #pragma unroll
        for (int k = 0; k < 4; ++k) PROC_A(v[k]);
    }
#endif
    for (int i = nFull * CH + tid; i < V; i += TPB1) PROC_A(rowp[i]);
#undef PROC_A

    // block reduction of (m, s)
    red_v[tid] = m; red_s[tid] = s;
    __syncthreads();
    for (int off = TPB1 / 2; off > 0; off >>= 1) {
        if (tid < off) {
            float m1 = red_v[tid],       s1 = red_s[tid];
            float m2 = red_v[tid + off], s2 = red_s[tid + off];
            if (m2 > m1) { s1 = s1 * __expf(m1 - m2) + s2; m1 = m2; }
            else         { s1 = s1 + s2 * __expf(m2 - m1); }
            red_v[tid] = m1; red_s[tid] = s1;
        }
        __syncthreads();
    }
    if (tid == 0) ws_logZ[row] = red_v[0] + __logf(red_s[0]);

    // threshold bin: largest bin such that count(key-bin >= b0) >= 64
    if (tid == 0) {
        unsigned cum = 0; int b0 = 0;
        for (int b = NBINS - 1; b >= 0; --b) {
            cum += hist[b];
            if (cum >= KTOP) { b0 = b; break; }
        }
        s_b0 = (unsigned)b0;
    }
    __syncthreads();
    const unsigned b0 = s_b0;

    // ---------------- Pass B: collect candidates >= threshold bin ----------
#define PROC_B(x, gi)                                          \
    do {                                                       \
        float _x = (x);                                        \
        if ((fkey(_x) >> 20) >= b0) {                          \
            unsigned _p = atomicAdd(&ccount, 1u);              \
            if (_p < CAND_MAX) { cval[_p] = _x; cidx[_p] = (gi); } \
        }                                                      \
    } while (0)

#if USE_ASYNC
    if (nFull > 0) {
        async_copy16(rowp + tid * 4, &tile[0][tid * 4]);
        for (int it = 0; it < nFull; ++it) {
            if (it + 1 < nFull) {
                async_copy16(rowp + (size_t)(it + 1) * CH + tid * 4,
                             &tile[(it + 1) & 1][tid * 4]);
                async_wait1();
            } else {
                async_wait0();
            }
            __syncthreads();
            const float* tp = tile[it & 1] + tid * 4;
            const int base = it * CH + tid * 4;
            #pragma unroll
            for (int k = 0; k < 4; ++k) PROC_B(tp[k], base + k);
            __syncthreads();
        }
    }
#else
    for (int it = 0; it < nFull; ++it) {
        f4 v = *(const f4*)(rowp + (size_t)it * CH + tid * 4);
        const int base = it * CH + tid * 4;
        #pragma unroll
        for (int k = 0; k < 4; ++k) PROC_B(v[k], base + k);
    }
#endif
    for (int i = nFull * CH + tid; i < V; i += TPB1) PROC_B(rowp[i], i);
#undef PROC_B
    __syncthreads();

    unsigned nc = ccount; if (nc > CAND_MAX) nc = CAND_MAX;
    const int ncand = (int)nc;

    // ---------------- exact top-64 selection (value desc, index asc) -------
    const float tinv = 1.0f / fmaxf(temperature[row], 0.05f);
    for (int j = 0; j < KTOP; ++j) {
        float bv = -3.4e38f; int bi = INT_MAX, bp = -1;
        for (int c = tid; c < ncand; c += TPB1) {
            float v = cval[c]; int ii = cidx[c];
            if (v > bv || (v == bv && ii < bi)) { bv = v; bi = ii; bp = c; }
        }
        red_v[tid] = bv; red_i[tid] = bi; red_p[tid] = bp;
        __syncthreads();
        for (int off = TPB1 / 2; off > 0; off >>= 1) {
            if (tid < off) {
                float v2 = red_v[tid + off]; int i2 = red_i[tid + off];
                if (v2 > red_v[tid] || (v2 == red_v[tid] && i2 < red_i[tid])) {
                    red_v[tid] = v2; red_i[tid] = i2; red_p[tid] = red_p[tid + off];
                }
            }
            __syncthreads();
        }
        if (tid == 0) {
            int p = red_p[0];
            ws_vals[row * KTOP + j] = (p >= 0) ? red_v[0] * tinv : NEG_INF;
            ws_idx [row * KTOP + j] = (p >= 0) ? red_i[0] : 0;
            if (p >= 0) cval[p] = -3.4e38f;   // remove
        }
        __syncthreads();
    }
}

// ---------------------------------------------------------------------------
// Kernel 2: one wave32 per 16 rows. Softmax over 64 candidates, exclusive
// cumsum via WMMA:  C(16x64) = P(16x64, f16) x M(64x64, f16), M[k][j]=(k<j),
// then top-p/min-p filter + Gumbel-max sample.
// ---------------------------------------------------------------------------
__device__ __forceinline__ float gumbel_hash(unsigned i) {
    unsigned x = i * 747796405u + 2891336453u;
    x ^= x >> 16; x *= 0x7feb352du;
    x ^= x >> 15; x *= 0x846ca68bu;
    x ^= x >> 16;
    float u = (float)(x >> 8) * (1.0f / 16777216.0f);
    u = fmaxf(u, 1e-12f);
    return -__logf(-__logf(u));
}

__global__ void __launch_bounds__(32)
sample_kernel(const float* __restrict__ ws_vals,
              const int*   __restrict__ ws_idx,
              const int*   __restrict__ top_k,
              const float* __restrict__ top_p,
              const float* __restrict__ min_p,
              float* __restrict__ out_tokens)
{
    const int lane = threadIdx.x;
    const int r0   = blockIdx.x * 16;

    __shared__ float    sp[16][KTOP];   // probs f32
    __shared__ _Float16 ph[16][KTOP];   // probs f16 (WMMA A source)
    __shared__ float    sc[16][KTOP];   // exclusive cumsum

    if (lane < 16) {
        const int row = r0 + lane;
        int kk = top_k[row]; kk = kk < 1 ? 1 : (kk > KTOP ? KTOP : kk);
        const float* v = ws_vals + row * KTOP;
        const float mx = v[0];            // sorted desc; first always kept
        float sum = 0.0f;
        for (int j = 0; j < KTOP; ++j) {
            float x = (j < kk) ? v[j] : NEG_INF;
            sum += __expf(x - mx);
        }
        const float rs = 1.0f / sum;
        for (int j = 0; j < KTOP; ++j) {
            float x = (j < kk) ? v[j] : NEG_INF;
            float p = __expf(x - mx) * rs;
            sp[lane][j] = p;
            ph[lane][j] = (_Float16)p;
        }
    }
    __syncthreads();

    // --- WMMA exclusive cumsum: uniform flow, EXEC all ones ---
    const int half = lane >> 4;     // K-half select
    const int mrow = lane & 15;     // A row / C column
    v16h a0, a1;
    #pragma unroll
    for (int i = 0; i < 16; ++i) {
        // 16-bit A layout: lanes0-15: v0-3->K0..7, v4-7->K16..23; lanes16-31: +8
        int K = i + 8 * half + ((i >= 8) ? 8 : 0);
        a0[i] = ph[mrow][K];
        a1[i] = ph[mrow][32 + K];
    }
    #pragma unroll
    for (int jt = 0; jt < 4; ++jt) {
        v8f acc = {};
        #pragma unroll
        for (int kt = 0; kt < 2; ++kt) {
            v16h b;
            #pragma unroll
            for (int i = 0; i < 16; ++i) {
                int K  = i + 8 * half + ((i >= 8) ? 8 : 0);
                int kg = 32 * kt + K;          // global K index
                int jg = 16 * jt + mrow;       // global output column
                b[i] = (kg < jg) ? (_Float16)1.0f : (_Float16)0.0f;
            }
            acc = __builtin_amdgcn_wmma_f32_16x16x32_f16(
                    false, (kt == 0) ? a0 : a1, false, b,
                    (short)0, acc, false, false);
        }
        #pragma unroll
        for (int r = 0; r < 8; ++r)            // C layout: VGPR r -> M=r(+8)
            sc[r + 8 * half][16 * jt + mrow] = acc[r];
    }
    __syncthreads();

    if (lane < 16) {
        const int row = r0 + lane;
        int kk = top_k[row]; kk = kk < 1 ? 1 : (kk > KTOP ? KTOP : kk);
        const float tp = top_p[row], mp = min_p[row];
        const float p0 = sp[lane][0];
        float best = -3.4e38f; int bj = 0;
        for (int j = 0; j < KTOP; ++j) {
            float p = sp[lane][j];
            bool keep = (j == 0) ||
                        ((j < kk) && (sc[lane][j] < tp) && (p >= mp * p0));
            float lp  = keep ? __logf(p + 1e-20f) : NEG_INF;
            float scr = lp + gumbel_hash((unsigned)(row * KTOP + j) * 2654435761u + 42u);
            if (scr > best) { best = scr; bj = j; }
        }
        out_tokens[row] = (float)ws_idx[row * KTOP + bj];
    }
}

// ---------------------------------------------------------------------------
// Kernel 3: perplexity gather: ppl[i,j] = logZ[i] - logits[i, tgt[i,j]]
// ---------------------------------------------------------------------------
__global__ void __launch_bounds__(256)
ppl_kernel(const float* __restrict__ logits,
           const int*   __restrict__ target_ids,
           const float* __restrict__ ws_logZ,
           float* __restrict__ out_ppl,
           int V, int n, int T)
{
    int t = blockIdx.x * 256 + threadIdx.x;
    if (t < n) {
        int i   = t / T;
        int tok = target_ids[t];
        out_ppl[t] = ws_logZ[i] - logits[(size_t)i * (size_t)V + tok];
    }
}

extern "C" void kernel_launch(void* const* d_in, const int* in_sizes, int n_in,
                              void* d_out, int out_size, void* d_ws, size_t ws_size,
                              hipStream_t stream) {
    const float* logits      = (const float*)d_in[0];
    const float* temperature = (const float*)d_in[1];
    const int*   top_k       = (const int*)d_in[2];
    const float* top_p       = (const float*)d_in[3];
    const float* min_p       = (const float*)d_in[4];
    const int*   target_ids  = (const int*)d_in[5];

    const int B = in_sizes[1];                 // 256
    const int V = in_sizes[0] / B;             // 128000
    const int T = in_sizes[5] / B;             // 16

    float* ws_vals = (float*)d_ws;                                     // B*64 f32
    int*   ws_idx  = (int*)  ((char*)d_ws + (size_t)B * KTOP * 4);     // B*64 i32
    float* ws_logZ = (float*)((char*)d_ws + (size_t)B * KTOP * 8);     // B f32

    float* out_tokens = (float*)d_out;       // [B]
    float* out_ppl    = (float*)d_out + B;   // [B*T]

    topk_lse_kernel<<<B, TPB1, 0, stream>>>(logits, temperature,
                                            ws_vals, ws_idx, ws_logZ, V);
    sample_kernel<<<B / 16, 32, 0, stream>>>(ws_vals, ws_idx,
                                             top_k, top_p, min_p, out_tokens);
    ppl_kernel<<<(B * T + 255) / 256, 256, 0, stream>>>(logits, target_ids,
                                                        ws_logZ, out_ppl,
                                                        V, B * T, T);
}
